// SegSampling_34076270527056
// MI455X (gfx1250) — matrix operand order repaired and checked by reading, not compiled, not run on hardware
//
#include <hip/hip_runtime.h>
#include <hip/hip_bf16.h>
#include <math.h>

typedef __attribute__((ext_vector_type(16))) _Float16 v16h;
typedef __attribute__((ext_vector_type(8)))  float    v8f;

// ---------------- problem constants ----------------
#define BB 1
#define NN 6
#define CC 256
#define GG 4
#define PP 8
#define LL 4
#define P1C 4
#define P2C 2
#define HQ 50
#define WQ 50
#define QQ (HQ*WQ)        // 2500
#define CGC 64
#define IMG_H 512.0f
#define IMG_W 1408.0f
#define EPSV 1e-5f

// pc_range
#define LOX (-51.2f)
#define LOY (-51.2f)
#define LOZ (-5.0f)
#define SPX 102.4f
#define SPY 102.4f
#define SPZ 8.0f

// ---------------------------------------------------------------------------
// WMMA GEMM:  Out[o,q] = sum_k W[o,k] * X[k,q] + bias[o]
// KT (=C=256) and NT (=Q=2500) are compile-time so all strided accesses fold
// into instruction immediate offsets (no address math inside the K-loop).
// One wave computes a 16x64 strip: A fragment reused across 4 WMMAs per K-step.
// REQUIREMENT (guaranteed by launches): O is a multiple of 16.
// ---------------------------------------------------------------------------
template<int KT, int NT>
__global__ void __launch_bounds__(128)
gemm_wmma_f16(const float* __restrict__ W, const float* __restrict__ bias,
              const float* __restrict__ X, float* __restrict__ Out, int O)
{
    const int lane   = threadIdx.x & 31;
    const int waveId = threadIdx.x >> 5;
    const int wavesPerBlock = blockDim.x >> 5;
    const int tile = blockIdx.x * wavesPerBlock + waveId;

    const int ntn = (NT + 63) >> 6;       // 64-wide N strips
    const int tm  = tile / ntn;
    const int tn  = tile % ntn;
    if (tm * 16 >= O) return;

    const int m     = lane & 15;          // A row within tile (both lane halves)
    const int half  = lane >> 4;          // 0: low K-half lanes, 1: high
    const int qbase = tn * 64 + (lane & 15);

    // loop-invariant clamped columns -> one fixed base pointer per fragment
    const int q0 = (qbase      < NT) ? qbase      : NT - 1;
    const int q1 = (qbase + 16 < NT) ? qbase + 16 : NT - 1;
    const int q2 = (qbase + 32 < NT) ? qbase + 32 : NT - 1;
    const int q3 = (qbase + 48 < NT) ? qbase + 48 : NT - 1;
    const float* __restrict__ xb0 = X + (size_t)half * 16 * NT + q0;
    const float* __restrict__ xb1 = X + (size_t)half * 16 * NT + q1;
    const float* __restrict__ xb2 = X + (size_t)half * 16 * NT + q2;
    const float* __restrict__ xb3 = X + (size_t)half * 16 * NT + q3;
    // A base: row tm*16+m, element e -> k = kb + half*16 + e
    const float* __restrict__ ab = W + (size_t)(tm * 16 + m) * KT + half * 16;

    v8f acc0 = {}, acc1 = {}, acc2 = {}, acc3 = {};

    for (int kb = 0; kb < KT; kb += 32) {
        v16h a;
#pragma unroll
        for (int e = 0; e < 8; ++e) {
            a[e]     = (_Float16)ab[kb + e];
            a[e + 8] = (_Float16)ab[kb + e + 8];
        }
        v16h b0, b1, b2, b3;
#pragma unroll
        for (int e = 0; e < 16; ++e) {
            const size_t r = (size_t)(kb + e) * NT;  // constant stride -> imm offset
            b0[e] = (_Float16)xb0[r];
            b1[e] = (_Float16)xb1[r];
            b2[e] = (_Float16)xb2[r];
            b3[e] = (_Float16)xb3[r];
        }
        // (neg_a, A, neg_b, B, c_mod, C, reuse_a, reuse_b)
        acc0 = __builtin_amdgcn_wmma_f32_16x16x32_f16(false, a, false, b0, (short)0, acc0, false, false);
        acc1 = __builtin_amdgcn_wmma_f32_16x16x32_f16(false, a, false, b1, (short)0, acc1, false, false);
        acc2 = __builtin_amdgcn_wmma_f32_16x16x32_f16(false, a, false, b2, (short)0, acc2, false, false);
        acc3 = __builtin_amdgcn_wmma_f32_16x16x32_f16(false, a, false, b3, (short)0, acc3, false, false);
    }

    // bias per output row held by this lane: o = tm*16 + v + half*8
    float bv[8];
#pragma unroll
    for (int v = 0; v < 8; ++v) bv[v] = bias[tm * 16 + v + half * 8];

    // store: lane holds D[v] at row (tm*16 + v + half*8), col q (O%16==0 -> no row guard)
#pragma unroll
    for (int s = 0; s < 4; ++s) {
        const int q = qbase + s * 16;
        if (q < NT) {
            const v8f& acc = (s == 0) ? acc0 : (s == 1) ? acc1 : (s == 2) ? acc2 : acc3;
            float* orow = Out + (size_t)(tm * 16 + half * 8) * NT + q;
#pragma unroll
            for (int v = 0; v < 8; ++v) {
                orow[(size_t)v * NT] = acc[v] + bv[v];   // constant stride -> imm offset
            }
        }
    }
}

// ---------------------------------------------------------------------------
// Sampling kernel: one wave32 per (q,g,p). Wave-uniform projection math;
// each lane accumulates channels `lane` and `lane+32` of the 64-channel group.
// ---------------------------------------------------------------------------
__device__ __forceinline__ void sample_level(const float* __restrict__ f,
                                             int n, int g, int lane,
                                             float u, float v, int H, int W,
                                             float sw, float& a0, float& a1)
{
    const float x = u * (float)W - 0.5f;
    const float y = v * (float)H - 0.5f;
    const float xf = floorf(x), yf = floorf(y);
    const int   x0 = (int)xf,   y0 = (int)yf;
    const float wx = x - xf,    wy = y - yf;

    const int HWl = H * W;
    const float* base0 = f + ((size_t)(n * CC + g * CGC) + lane) * (size_t)HWl;
    const float* base1 = base0 + (size_t)32 * HWl;

    const float w00 = (1.0f - wx) * (1.0f - wy);
    const float w10 = wx * (1.0f - wy);
    const float w01 = (1.0f - wx) * wy;
    const float w11 = wx * wy;

#pragma unroll
    for (int c = 0; c < 4; ++c) {
        const int xi = x0 + (c & 1);
        const int yi = y0 + (c >> 1);
        if (xi < 0 || xi > W - 1 || yi < 0 || yi > H - 1) continue;
        const float wt = sw * ((c == 0) ? w00 : (c == 1) ? w10 : (c == 2) ? w01 : w11);
        const int off = yi * W + xi;
        a0 += wt * base0[off];
        a1 += wt * base1[off];
    }
}

__global__ void __launch_bounds__(256)
sample_kernel(const float* __restrict__ OFF,   // [96, Q]
              const float* __restrict__ SWB,   // [128, Q]
              const float* __restrict__ rp,    // [1,P1,HQ,WQ,3]
              const float* __restrict__ l2i,   // [1,N,4,4]
              const float* __restrict__ f0, const float* __restrict__ f1,
              const float* __restrict__ f2, const float* __restrict__ f3,
              float* __restrict__ out)         // [Q,G,P,CG]
{
    const int lane = threadIdx.x & 31;
    const int wave = threadIdx.x >> 5;
    const int idx  = blockIdx.x * (blockDim.x >> 5) + wave;  // q*G*P + g*P + p
    if (idx >= QQ * GG * PP) return;

    const int p = idx % PP;
    const int g = (idx / PP) % GG;
    const int q = idx / (PP * GG);
    const int h = q / WQ, w = q % WQ;
    const int p1 = p / P2C;   // p = p1*P2 + p2

    // learned offsets: sigmoid(conv) - 0.5, channel o = (g*P+p)*3 + d
    const int o0 = (g * PP + p) * 3;
    const float ox = 1.0f / (1.0f + __expf(-OFF[(size_t)(o0 + 0) * QQ + q])) - 0.5f;
    const float oy = 1.0f / (1.0f + __expf(-OFF[(size_t)(o0 + 1) * QQ + q])) - 0.5f;
    const float oz = 1.0f / (1.0f + __expf(-OFF[(size_t)(o0 + 2) * QQ + q])) - 0.5f;

    // reference point (per p1) unnormalized to pc_range
    const float* rpp = rp + ((size_t)(p1 * HQ + h) * WQ + w) * 3;
    const float px = rpp[0] * SPX + LOX + ox;
    const float py = rpp[1] * SPY + LOY + oy;
    const float pz = rpp[2] * SPZ + LOZ + oz;

    // scale weights: softmax over L, channel o = (g*P+p)*L + l
    const int s0 = (g * PP + p) * LL;
    float s[LL];
    float smax = -1e30f;
#pragma unroll
    for (int l = 0; l < LL; ++l) {
        s[l] = SWB[(size_t)(s0 + l) * QQ + q];
        smax = fmaxf(smax, s[l]);
    }
    float ssum = 0.0f;
#pragma unroll
    for (int l = 0; l < LL; ++l) { s[l] = __expf(s[l] - smax); ssum += s[l]; }
    const float sinv = 1.0f / ssum;
#pragma unroll
    for (int l = 0; l < LL; ++l) s[l] *= sinv;

    float acc0 = 0.0f, acc1 = 0.0f;

#pragma unroll
    for (int n = 0; n < NN; ++n) {
        const float* M = l2i + n * 16;
        const float cx = M[0]  * px + M[1]  * py + M[2]  * pz + M[3];
        const float cy = M[4]  * px + M[5]  * py + M[6]  * pz + M[7];
        const float cz = M[8]  * px + M[9]  * py + M[10] * pz + M[11];
        const float zc = fmaxf(cz, EPSV);
        const float u  = cx / zc * (1.0f / IMG_W);
        const float v  = cy / zc * (1.0f / IMG_H);
        const bool valid = (cz > EPSV) && (u > 0.0f) && (u < 1.0f)
                                       && (v > 0.0f) && (v < 1.0f);
        if (!valid) continue;   // wave-uniform branch

        sample_level(f0, n, g, lane, u, v, 64, 176, s[0], acc0, acc1);
        sample_level(f1, n, g, lane, u, v, 32,  88, s[1], acc0, acc1);
        sample_level(f2, n, g, lane, u, v, 16,  44, s[2], acc0, acc1);
        sample_level(f3, n, g, lane, u, v,  8,  22, s[3], acc0, acc1);
    }

    float* op = out + (size_t)idx * CGC;
    op[lane]      = acc0;
    op[lane + 32] = acc1;
}

// ---------------------------------------------------------------------------
extern "C" void kernel_launch(void* const* d_in, const int* in_sizes, int n_in,
                              void* d_out, int out_size, void* d_ws, size_t ws_size,
                              hipStream_t stream)
{
    const float* query  = (const float*)d_in[0];   // [1,256,50,50] -> [256, 2500]
    const float* refpts = (const float*)d_in[1];   // [1,4,50,50,3]
    const float* l2i    = (const float*)d_in[2];   // [1,6,4,4]
    const float* w_off  = (const float*)d_in[3];   // [96,256]
    const float* b_off  = (const float*)d_in[4];   // [96]
    const float* w_sw   = (const float*)d_in[5];   // [128,256]
    const float* b_sw   = (const float*)d_in[6];   // [128]
    const float* f0     = (const float*)d_in[7];
    const float* f1     = (const float*)d_in[8];
    const float* f2     = (const float*)d_in[9];
    const float* f3     = (const float*)d_in[10];

    float* OFF = (float*)d_ws;            // 96 * 2500
    float* SWB = OFF + 96 * QQ;           // 128 * 2500  (total 2.24 MB)

    const int ntn = (QQ + 63) / 64;       // 40 N-strips of 64

    {   // offsets GEMM: 96x2500x256 -> 6 * 40 = 240 wave-strips
        const int tiles  = (96 / 16) * ntn;
        const int blocks = (tiles + 3) / 4;           // 4 waves / block
        gemm_wmma_f16<CC, QQ><<<blocks, 128, 0, stream>>>(w_off, b_off, query, OFF, 96);
    }
    {   // scale-weight GEMM: 128x2500x256 -> 8 * 40 = 320 wave-strips
        const int tiles  = (128 / 16) * ntn;
        const int blocks = (tiles + 3) / 4;
        gemm_wmma_f16<CC, QQ><<<blocks, 128, 0, stream>>>(w_sw, b_sw, query, SWB, 128);
    }
    {   // sampling: one wave per (q,g,p) = 80000 waves, 8 waves / block
        const int waves  = QQ * GG * PP;
        const int blocks = waves / 8;                 // 10000
        sample_kernel<<<blocks, 256, 0, stream>>>(OFF, SWB, refpts, l2i,
                                                  f0, f1, f2, f3, (float*)d_out);
    }
}
